// TimeSeriesPositionalRecommendation_27874337751080
// MI455X (gfx1250) — matrix operand order repaired and checked by reading, not compiled, not run on hardware
//
#include <hip/hip_runtime.h>

// ---------------------------------------------------------------------------
// MI455X (gfx1250). Dominant work: 3x per-chunk (A[1024x1024]@H) aggregations
// = 256MB of adjacency per pass from HBM -> bandwidth bound (~11us/pass at
// 23.3 TB/s). bf16 WMMA (v_wmma_f32_16x16x32_bf16) keeps the matmul at the
// bandwidth roofline. A-slabs are moved by the Tensor Data Mover into LDS
// (double buffered, TENSORcnt-synchronized); waves only build fragments and
// issue WMMAs. Per-layer dense projection (@W^T + b, relu) fused in LDS.
// ---------------------------------------------------------------------------

typedef __attribute__((ext_vector_type(16))) __bf16 v16bf;
typedef __attribute__((ext_vector_type(8)))  float  v8f;
typedef __attribute__((ext_vector_type(4)))  unsigned v4u;
typedef __attribute__((ext_vector_type(4)))  int      v4i;
typedef __attribute__((ext_vector_type(8)))  int      v8i;

#define NND   1024   // nodes per chunk
#define NCH   64     // chunks
#define HID   64

#if __has_builtin(__builtin_amdgcn_tensor_load_to_lds)
#define USE_TDM 1
#else
#define USE_TDM 0
#endif

// LDS pitches (elements). 36 f32 = 144B rows; 72 bf16 = 144B rows: every
// fragment chunk is 16B-aligned and row stride is an odd multiple of 16B.
#define AST_F 36
#define BST   72
#define WST   72
#define SST   72
#define KSTEPS (NND / 32)

__device__ __forceinline__ v8f zero8() {
  v8f z;
#pragma unroll
  for (int i = 0; i < 8; ++i) z[i] = 0.0f;
  return z;
}

// A fragment (16x32 bf16), ISA 7.12.2 layout: lane&15 = M, half=lane>>4,
// VGPR r pair k0 = (r>>2)*16 + half*8 + (r&3)*2. Per lane this is two
// contiguous 16B runs at row + half*8 and row + half*8 + 16.
__device__ __forceinline__ v16bf load_a_frag_bf16(const __bf16* base, int stride, int lane) {
  union { v16bf v; uint4 q[2]; } f;
  const int m = lane & 15, half = lane >> 4;
  const __bf16* row = base + m * stride + half * 8;
  f.q[0] = *(const uint4*)(row);
  f.q[1] = *(const uint4*)(row + 16);
  return f.v;
}

// Same fragment out of an f32 slab (TDM-staged adjacency), converting to bf16.
__device__ __forceinline__ v16bf load_a_frag_f32cvt(const float* base, int stride, int lane) {
  union { v16bf v; __bf16 h[16]; } f;
  const int m = lane & 15, half = lane >> 4;
  const float* row = base + m * stride + half * 8;
  const float4 c0 = *(const float4*)(row);
  const float4 c1 = *(const float4*)(row + 4);
  const float4 c2 = *(const float4*)(row + 16);
  const float4 c3 = *(const float4*)(row + 20);
  f.h[0]  = (__bf16)c0.x; f.h[1]  = (__bf16)c0.y; f.h[2]  = (__bf16)c0.z; f.h[3]  = (__bf16)c0.w;
  f.h[4]  = (__bf16)c1.x; f.h[5]  = (__bf16)c1.y; f.h[6]  = (__bf16)c1.z; f.h[7]  = (__bf16)c1.w;
  f.h[8]  = (__bf16)c2.x; f.h[9]  = (__bf16)c2.y; f.h[10] = (__bf16)c2.z; f.h[11] = (__bf16)c2.w;
  f.h[12] = (__bf16)c3.x; f.h[13] = (__bf16)c3.y; f.h[14] = (__bf16)c3.z; f.h[15] = (__bf16)c3.w;
  return f.v;
}

// B fragment (32x16 bf16) from a k-major slab (element (k,n) at base[k*stride+n]).
// Lane&15 = N, half=lane>>4; element s holds k = half*8 + s (s<8) and
// k = 16 + half*8 + (s-8) (s>=8), matching the A fragment.
__device__ __forceinline__ v16bf load_b_frag(const __bf16* base, int stride, int ncol0, int lane) {
  union { v16bf v; __bf16 h[16]; } f;
  const int n = lane & 15, half = lane >> 4;
  const __bf16* p = base + (half * 8) * stride + ncol0 + n;
#pragma unroll
  for (int s = 0; s < 8; ++s) {
    f.h[s]     = p[s * stride];
    f.h[8 + s] = p[(16 + s) * stride];
  }
  return f.v;
}

__device__ __forceinline__ v8f wmma_bf16(v16bf a, v16bf b, v8f c) {
  return __builtin_amdgcn_wmma_f32_16x16x32_bf16(false, a, false, b, (short)0, c, false, false);
}

#if USE_TDM
// Issue a TDM 2D tile load: 32 f32 x 128 rows, row stride 1024 f32, into LDS
// with a 4-dword pad every 32 dwords (-> 36-dword LDS pitch). D# layout per
// CDNA5 ISA (group0 128b, group1 256b, g2/g3 zero).
// This toolchain's builtin takes 6 args: (g0 v4u, g1 v8i, g2 v4i, g3 v4i,
// extra v8i, cpol i32).
__device__ __forceinline__ void tdm_load_a(const float* gsrc, unsigned lds_off) {
  const unsigned long long ga = (unsigned long long)(uintptr_t)gsrc;
  v4u g0;
  g0[0] = 1u;                                        // count=1, user mode
  g0[1] = lds_off;                                   // LDS byte address
  g0[2] = (unsigned)(ga & 0xFFFFFFFFu);              // global_addr[31:0]
  g0[3] = (unsigned)((ga >> 32) & 0x01FFFFFFu) | 0x80000000u; // [56:32] | type=2
  v8i g1;
  g1[0] = (int)((2u << 16)        // data_size = 4B
              | (1u << 20)        // pad_enable
              | (4u << 22)        // pad_interval: 32 dwords
              | (3u << 25));      // pad_amount: 4 dwords -> 36-dword pitch
  g1[1] = (int)(32u << 16);       // tensor_dim0 = 32 (lo16); atomic_barrier=0
  g1[2] = (int)(128u << 16);      // tensor_dim0 hi=0 | tensor_dim1 lo16 = 128
  g1[3] = (int)(32u << 16);       // tensor_dim1 hi=0 | tile_dim0 = 32
  g1[4] = 128;                    // tile_dim1 = 128, tile_dim2 = 0
  g1[5] = 1024;                   // tensor_dim0_stride = 1024 elements
  g1[6] = 0;
  g1[7] = 0;
  const v4i z4 = {0, 0, 0, 0};
  const v8i z8 = {0, 0, 0, 0, 0, 0, 0, 0};
  __builtin_amdgcn_tensor_load_to_lds(g0, g1, z4, z4, z8, 0);
}
#endif

// ---------------------------------------------------------------------------
// Fused GCN layer:  Hout = act( (A @ Hin) @ W^T + b )
// grid = (8 row-stripes, 64 chunks), block = 256 (8 waves x 16 rows).
// ---------------------------------------------------------------------------
template <int CIN, bool INTRANS, bool RELU, bool OUTTRANS>
__global__ __launch_bounds__(256) void gcn_layer(
    const float* __restrict__ A, const void* __restrict__ HinV,
    const float* __restrict__ W, const float* __restrict__ bias,
    __bf16* __restrict__ Hout)
{
  constexpr int SCT = (CIN <= 16) ? 1 : 4;   // S column tiles from aggregation
  constexpr int SK  = (CIN <= 16) ? 32 : 64; // padded K of the dense stage
  // LDS carve-out: A double buffer (f32, TDM dest) aliased later by S slab.
  constexpr int A_BYTES = 2 * 128 * AST_F * 4;          // 36864
  constexpr int B_BYTES = 2 * 32 * BST * 2;             // 9216
  constexpr int W_BYTES = 64 * WST * 2;                 // 9216
  __shared__ __align__(16) char smem[A_BYTES + B_BYTES + W_BYTES];
  float*  sAf = (float*)smem;                     // [2][128][36] f32
  __bf16* sS  = (__bf16*)smem;                    // [128][72] bf16 (aliases sAf)
  __bf16* sB  = (__bf16*)(smem + A_BYTES);        // [2][32][72] bf16, k-major
  __bf16* sW  = (__bf16*)(smem + A_BYTES + B_BYTES); // [SK][72] bf16, k-major

  const int t = threadIdx.x;
  const int lane = t & 31, wave = t >> 5;
  const int waveU = __builtin_amdgcn_readfirstlane(wave); // force scalar branch
  const int b = blockIdx.y;
  const int row0 = blockIdx.x * 128;
  const size_t abase = ((size_t)b << 20) + (size_t)row0 * NND;

  // ---- stage W^T k-major (zero padded to SK rows)
  for (int e = t; e < SK * HID; e += 256) {
    const int k = e >> 6, n = e & 63;
    sW[k * WST + n] = (k < CIN) ? (__bf16)W[n * CIN + k] : (__bf16)0.0f;
  }

  // B-slab staging helper
  auto stageB = [&](int kb, __bf16* dst) {
    if constexpr (INTRANS) {
      const __bf16* HT = (const __bf16*)HinV + (size_t)b * HID * NND;
      const int n = t >> 2, kq = (t & 3) * 8;
      const __bf16* src = HT + (size_t)n * NND + kb * 32 + kq;
#pragma unroll
      for (int i = 0; i < 8; ++i) dst[(kq + i) * BST + n] = src[i];
    } else {
      const float* X = (const float*)HinV + (size_t)b * NND * CIN;
      for (int e = t; e < 32 * 16; e += 256) {
        const int k = e >> 4, n = e & 15;
        dst[k * BST + n] = (n < CIN) ? (__bf16)X[(kb * 32 + k) * CIN + n]
                                     : (__bf16)0.0f;
      }
    }
  };

#if !USE_TDM
  auto stageA = [&](int kb, float* dst) {
#pragma unroll
    for (int i = 0; i < 4; ++i) {
      const int g = i * 256 + t;
      const int row = g >> 3, c4 = g & 7;
      const float* src = A + abase + (size_t)row * NND + kb * 32 + c4 * 4;
      const float4 v = *(const float4*)src;
      float* d = &dst[row * AST_F + c4 * 4];
      d[0] = v.x; d[1] = v.y; d[2] = v.z; d[3] = v.w;
      if (i == 0 && kb + 1 < KSTEPS) __builtin_prefetch(src + 32, 0, 0);
    }
  };
#endif

  const unsigned aOff0 = (unsigned)(uintptr_t)(sAf);
  const unsigned aOff1 = (unsigned)(uintptr_t)(sAf + 128 * AST_F);

  // ---- prologue: fill buffer 0
#if USE_TDM
  if (waveU == 0) tdm_load_a(A + abase, aOff0);
#else
  stageA(0, sAf);
#endif
  stageB(0, sB);
#if USE_TDM
  if (waveU == 0) __builtin_amdgcn_s_wait_tensorcnt(0);
#endif
  __syncthreads();

  v8f accS[SCT];
#pragma unroll
  for (int j = 0; j < SCT; ++j) accS[j] = zero8();

  // ---- main K loop: TDM prefetches slab k+1 while waves WMMA slab k
  for (int kb = 0; kb < KSTEPS; ++kb) {
    const int cur = kb & 1, nxt = cur ^ 1;
    float*  aCur = sAf + cur * (128 * AST_F);
    __bf16* bCur = sB + cur * (32 * BST);
    if (kb + 1 < KSTEPS) {
#if USE_TDM
      if (waveU == 0)
        tdm_load_a(A + abase + (kb + 1) * 32, nxt ? aOff1 : aOff0);
#else
      stageA(kb + 1, sAf + nxt * (128 * AST_F));
#endif
      stageB(kb + 1, sB + nxt * (32 * BST));
    }
    const v16bf af = load_a_frag_f32cvt(aCur + wave * 16 * AST_F, AST_F, lane);
#pragma unroll
    for (int j = 0; j < SCT; ++j) {
      const v16bf bf = load_b_frag(bCur, BST, j * 16, lane);
      accS[j] = wmma_bf16(af, bf, accS[j]);
    }
#if USE_TDM
    if (kb + 1 < KSTEPS && waveU == 0) __builtin_amdgcn_s_wait_tensorcnt(0);
#endif
    __syncthreads();
  }

  // ---- A buffers dead; reuse as S slab. Zero (covers stage-1 K padding).
  for (int e = t; e < 128 * SST; e += 256) sS[e] = (__bf16)0.0f;
  __syncthreads();
  {
    const int nl = lane & 15, half = lane >> 4;
#pragma unroll
    for (int j = 0; j < SCT; ++j)
#pragma unroll
      for (int r = 0; r < 8; ++r)
        sS[(wave * 16 + r + 8 * half) * SST + j * 16 + nl] = (__bf16)accS[j][r];
  }
  __syncthreads();

  // ---- fused dense: OUT = act(S @ W^T + b), 16x64 per wave
#pragma unroll
  for (int jo = 0; jo < 4; ++jo) {
    v8f acc = zero8();
#pragma unroll
    for (int kt = 0; kt < SK / 32; ++kt) {
      const v16bf af = load_a_frag_bf16(&sS[wave * 16 * SST + kt * 32], SST, lane);
      const v16bf bf = load_b_frag(&sW[kt * 32 * WST], WST, jo * 16, lane);
      acc = wmma_bf16(af, bf, acc);
    }
    const int nl = lane & 15, half = lane >> 4;
    const float bb = bias[jo * 16 + nl];
    if constexpr (OUTTRANS) {
      union { uint4 q; __bf16 h[8]; } pk;
#pragma unroll
      for (int r = 0; r < 8; ++r) {
        float v = acc[r] + bb;
        if constexpr (RELU) v = v > 0.0f ? v : 0.0f;
        pk.h[r] = (__bf16)v;
      }
      const int f = jo * 16 + nl;
      const size_t idx = ((size_t)b * HID + f) * NND + row0 + wave * 16 + 8 * half;
      *(uint4*)(Hout + idx) = pk.q;     // 16B store, 8 consecutive nodes
    } else {
#pragma unroll
      for (int r = 0; r < 8; ++r) {
        float v = acc[r] + bb;
        if constexpr (RELU) v = v > 0.0f ? v : 0.0f;
        const size_t node = (size_t)row0 + wave * 16 + r + 8 * half;
        Hout[((size_t)b * NND + node) * HID + jo * 16 + nl] = (__bf16)v;
      }
    }
  }
}

// ---------------------------------------------------------------------------
__global__ void chunk_mean(const __bf16* __restrict__ ne, float* __restrict__ cr) {
  const int b = blockIdx.x, f = threadIdx.x;   // 64 threads
  float s = 0.0f;
  const __bf16* p = ne + (size_t)b * NND * HID + f;
  for (int n = 0; n < NND; ++n) s += (float)p[(size_t)n * HID];
  cr[b * HID + f] = s * (1.0f / (float)NND);
}

// ---------------------------------------------------------------------------
// BiLSTM (T=64, H=32/dir) + time encoder + scorer bias precompute.
// threads [0,128) forward dir, [128,256) backward; one thread per gate row.
// ---------------------------------------------------------------------------
__global__ __launch_bounds__(256) void lstm_time(
    const float* __restrict__ cr,
    const float* __restrict__ wih_f, const float* __restrict__ whh_f,
    const float* __restrict__ bih_f, const float* __restrict__ bhh_f,
    const float* __restrict__ wih_b, const float* __restrict__ whh_b,
    const float* __restrict__ bih_b, const float* __restrict__ bhh_b,
    const float* __restrict__ te_w, const float* __restrict__ te_b,
    const float* __restrict__ ns1_w, const float* __restrict__ ns1_b,
    float* __restrict__ bias2)
{
  __shared__ float hbuf[2][32];
  __shared__ float zbuf[2][128];
  __shared__ float hsum[2][32];
  __shared__ float meanv[64];
  __shared__ float trepr[64];

  const int t = threadIdx.x;
  const int d = t >> 7, g = t & 127;
  const float* wih = d ? wih_b : wih_f;
  const float* whh = d ? whh_b : whh_f;
  const float bsum = d ? (bih_b[g] + bhh_b[g]) : (bih_f[g] + bhh_f[g]);
  float c = 0.0f;
  if (g < 32) { hbuf[d][g] = 0.0f; hsum[d][g] = 0.0f; }
  __syncthreads();

  for (int step = 0; step < NCH; ++step) {
    const int tf = d ? (NCH - 1 - step) : step;
    float z = bsum;
    const float* x = cr + tf * HID;
    for (int k = 0; k < HID; ++k) z += x[k] * wih[g * HID + k];
    for (int k = 0; k < 32; ++k) z += hbuf[d][k] * whh[g * 32 + k];
    zbuf[d][g] = z;
    __syncthreads();
    if (g < 32) {
      const float zi = zbuf[d][g],      zf = zbuf[d][32 + g];
      const float zg = zbuf[d][64 + g], zo = zbuf[d][96 + g];
      const float i  = 1.0f / (1.0f + __expf(-zi));
      const float f  = 1.0f / (1.0f + __expf(-zf));
      const float gg = tanhf(zg);
      const float o  = 1.0f / (1.0f + __expf(-zo));
      c = f * c + i * gg;
      const float h = o * tanhf(c);
      hbuf[d][g] = h;
      hsum[d][g] += h;
    }
    __syncthreads();
  }
  if (t < 32)       meanv[t] = hsum[0][t] * (1.0f / (float)NCH);
  else if (t < 64)  meanv[t] = hsum[1][t - 32] * (1.0f / (float)NCH);
  __syncthreads();
  if (t < 64) {
    float s = te_b[t];
    for (int k = 0; k < 64; ++k) s += meanv[k] * te_w[t * 64 + k];
    trepr[t] = s > 0.0f ? s : 0.0f;
  }
  __syncthreads();
  if (t < 64) {                       // bias2 = ns1_b + W2 @ time_repr
    float s = ns1_b[t];
    for (int k = 0; k < 64; ++k) s += trepr[k] * ns1_w[t * 128 + 64 + k];
    bias2[t] = s;
  }
}

// ---------------------------------------------------------------------------
// Node scorer: h = relu(target @ W1^T + bias2); scores = sigmoid(h @ ns2 + b)
// ---------------------------------------------------------------------------
__global__ __launch_bounds__(256) void scorer(
    const __bf16* __restrict__ ne, const int* __restrict__ pidx_p,
    const float* __restrict__ ns1_w, const float* __restrict__ bias2,
    const float* __restrict__ ns2_w, const float* __restrict__ ns2_b,
    float* __restrict__ out)
{
  constexpr int HST = 66;
  __shared__ __align__(16) __bf16 sW[64 * WST];  // W1^T, k-major bf16
  __shared__ float sH[128 * HST];                // relu(h) rows for this block

  const int t = threadIdx.x, lane = t & 31, wave = t >> 5;
  for (int e = t; e < 64 * 64; e += 256) {
    const int k = e >> 6, n = e & 63;
    sW[k * WST + n] = (__bf16)ns1_w[n * 128 + k];
  }
  __syncthreads();

  const int pidx = *pidx_p;
  const int row0 = blockIdx.x * 128 + wave * 16;
  const __bf16* tgt = ne + ((size_t)pidx * NND + row0) * HID;

#pragma unroll
  for (int jo = 0; jo < 4; ++jo) {
    v8f acc = zero8();
#pragma unroll
    for (int kt = 0; kt < 2; ++kt) {
      const v16bf af = load_a_frag_bf16(tgt + kt * 32, HID, lane);
      const v16bf bf = load_b_frag(&sW[kt * 32 * WST], WST, jo * 16, lane);
      acc = wmma_bf16(af, bf, acc);
    }
    const int nl = lane & 15, half = lane >> 4;
    const float bb = bias2[jo * 16 + nl];
#pragma unroll
    for (int r = 0; r < 8; ++r) {
      float v = acc[r] + bb;
      v = v > 0.0f ? v : 0.0f;
      sH[(wave * 16 + r + 8 * half) * HST + jo * 16 + nl] = v;
    }
  }
  __syncthreads();
  if (t < 128) {
    const float* hr = &sH[t * HST];
    float s = ns2_b[0];
    for (int k = 0; k < 64; ++k) s += hr[k] * ns2_w[k];
    out[blockIdx.x * 128 + t] = 1.0f / (1.0f + __expf(-s));
  }
}

// ---------------------------------------------------------------------------
extern "C" void kernel_launch(void* const* d_in, const int* in_sizes, int n_in,
                              void* d_out, int out_size, void* d_ws, size_t ws_size,
                              hipStream_t stream) {
  (void)in_sizes; (void)n_in; (void)out_size; (void)ws_size;
  const float* feat = (const float*)d_in[0];
  const float* adj  = (const float*)d_in[1];
  const float* gc1w = (const float*)d_in[2];
  const float* gc1b = (const float*)d_in[3];
  const float* gc2w = (const float*)d_in[4];
  const float* gc2b = (const float*)d_in[5];
  const float* gc3w = (const float*)d_in[6];
  const float* gc3b = (const float*)d_in[7];
  const float* wihf = (const float*)d_in[8];
  const float* whhf = (const float*)d_in[9];
  const float* bihf = (const float*)d_in[10];
  const float* bhhf = (const float*)d_in[11];
  const float* wihb = (const float*)d_in[12];
  const float* whhb = (const float*)d_in[13];
  const float* bihb = (const float*)d_in[14];
  const float* bhhb = (const float*)d_in[15];
  const float* tew  = (const float*)d_in[16];
  const float* teb  = (const float*)d_in[17];
  const float* ns1w = (const float*)d_in[18];
  const float* ns1b = (const float*)d_in[19];
  const float* ns2w = (const float*)d_in[20];
  const float* ns2b = (const float*)d_in[21];
  const int*   pidx = (const int*)d_in[23];   // d_in[22] = chunk_sizes (unused)

  char* ws = (char*)d_ws;
  const size_t layerB = (size_t)NCH * NND * HID;     // elements per layer buf
  __bf16* hT1     = (__bf16*)(ws);                   // feature-major bf16
  __bf16* hT2     = (__bf16*)(ws + layerB * 2);      // feature-major bf16
  __bf16* nodeemb = (__bf16*)(ws + layerB * 4);      // node-major bf16
  float*  crepr   = (float*)(ws + layerB * 6);
  float*  bias2   = (float*)(ws + layerB * 6 + NCH * HID * 4);

  dim3 grid(8, NCH), block(256);
  gcn_layer<3,  false, true,  true ><<<grid, block, 0, stream>>>(adj, feat, gc1w, gc1b, hT1);
  gcn_layer<64, true,  true,  true ><<<grid, block, 0, stream>>>(adj, hT1,  gc2w, gc2b, hT2);
  gcn_layer<64, true,  false, false><<<grid, block, 0, stream>>>(adj, hT2,  gc3w, gc3b, nodeemb);
  chunk_mean<<<NCH, HID, 0, stream>>>(nodeemb, crepr);
  lstm_time<<<1, 256, 0, stream>>>(crepr, wihf, whhf, bihf, bhhf,
                                   wihb, whhb, bihb, bhhb,
                                   tew, teb, ns1w, ns1b, bias2);
  scorer<<<8, 256, 0, stream>>>(nodeemb, pidx, ns1w, bias2, ns2w, ns2b, (float*)d_out);
}
